// BitNetMLP_22282290332379
// MI455X (gfx1250) — 8-bit matrix operand order repaired and checked by reading, compiled, not run on hardware
//
#include <hip/hip_runtime.h>

#define HID   4096
#define INTER 11008
#define MTOT  4096

typedef int v8i __attribute__((ext_vector_type(8)));

#if defined(__AMDGCN__) && __has_builtin(__builtin_amdgcn_global_load_async_to_lds_b128)
#define HAVE_ASYNC 1
#else
#define HAVE_ASYNC 0
#endif

#if HAVE_ASYNC
typedef int v4i_b __attribute__((vector_size(16)));
typedef __attribute__((address_space(1))) v4i_b gv4i;
typedef __attribute__((address_space(3))) v4i_b lv4i;

__device__ inline void async_cp_b128(const void* g, void* l) {
    __builtin_amdgcn_global_load_async_to_lds_b128(
        (gv4i*)(uintptr_t)g,
        (lv4i*)l,
        0, 0);
}

#if __has_builtin(__builtin_amdgcn_s_wait_asynccnt)
__device__ inline void wait_async_0()  { __builtin_amdgcn_s_wait_asynccnt(0); }
__device__ inline void wait_async_8()  { __builtin_amdgcn_s_wait_asynccnt(8); }
__device__ inline void wait_async_12() { __builtin_amdgcn_s_wait_asynccnt(12); }
#else
__device__ inline void wait_async_0()  { asm volatile("s_wait_asynccnt 0x0" ::: "memory"); }
__device__ inline void wait_async_8()  { asm volatile("s_wait_asynccnt 0x8" ::: "memory"); }
__device__ inline void wait_async_12() { asm volatile("s_wait_asynccnt 0xc" ::: "memory"); }
#endif
#endif

__global__ __launch_bounds__(256) void bitnet_maxabs_kernel(const float* __restrict__ w,
                                                            float* __restrict__ out) {
    __shared__ float red[256];
    int tid = threadIdx.x;
    float m = 0.f;
    for (int i = tid; i < 65536; i += 256) m = fmaxf(m, fabsf(w[i]));
    red[tid] = m;
    __syncthreads();
    for (int s = 128; s > 0; s >>= 1) {
        if (tid < s) red[tid] = fmaxf(red[tid], red[tid + s]);
        __syncthreads();
    }
    if (tid == 0) out[0] = red[0];
}

__device__ inline uint pack4_round_clamp(float4 f) {
    int q0 = min(127, max(-128, __float2int_rn(f.x)));
    int q1 = min(127, max(-128, __float2int_rn(f.y)));
    int q2 = min(127, max(-128, __float2int_rn(f.z)));
    int q3 = min(127, max(-128, __float2int_rn(f.w)));
    return (uint)(q0 & 0xff) | ((uint)(q1 & 0xff) << 8) |
           ((uint)(q2 & 0xff) << 16) | ((uint)(q3 & 0xff) << 24);
}

__device__ inline int tern_sign(float w) { return (w > 0.f) ? 1 : ((w < 0.f) ? -1 : 0); }

__device__ inline uint pack4_sign(float4 f) {
    int q0 = tern_sign(f.x), q1 = tern_sign(f.y), q2 = tern_sign(f.z), q3 = tern_sign(f.w);
    return (uint)(q0 & 0xff) | ((uint)(q1 & 0xff) << 8) |
           ((uint)(q2 & 0xff) << 16) | ((uint)(q3 & 0xff) << 24);
}

__global__ __launch_bounds__(256) void bitnet_quant_x_kernel(const float* __restrict__ x,
                                                             signed char* __restrict__ out) {
    size_t base = ((size_t)blockIdx.x * 256 + threadIdx.x) * 16;
    const float4* s = (const float4*)(x + base);
    uint4 p;
    p.x = pack4_round_clamp(s[0]);
    p.y = pack4_round_clamp(s[1]);
    p.z = pack4_round_clamp(s[2]);
    p.w = pack4_round_clamp(s[3]);
    *(uint4*)(out + base) = p;
}

__global__ __launch_bounds__(256) void bitnet_sign_kernel(const float* __restrict__ w,
                                                          signed char* __restrict__ out) {
    size_t base = ((size_t)blockIdx.x * 256 + threadIdx.x) * 16;
    const float4* s = (const float4*)(w + base);
    uint4 p;
    p.x = pack4_sign(s[0]);
    p.y = pack4_sign(s[1]);
    p.z = pack4_sign(s[2]);
    p.w = pack4_sign(s[3]);
    *(uint4*)(out + base) = p;
}

__device__ inline v8i load_a_frag(const uint* lds, int strideDw, int m16, int ksDw, int lane) {
    int row  = m16 + (lane & 15);
    int base = row * strideDw + ksDw + ((lane >> 4) << 2);
    v8i a;
#pragma unroll
    for (int j = 0; j < 2; ++j) {
        uint4 u = *(const uint4*)(lds + base + j * 8);
        a[4 * j]     = (int)u.x;
        a[4 * j + 1] = (int)u.y;
        a[4 * j + 2] = (int)u.z;
        a[4 * j + 3] = (int)u.w;
    }
    return a;
}

__device__ inline v8i load_b_frag(const uint* lds, int strideDw, int n16, int ksDw, int lane) {
    int col  = n16 + (lane & 15);
    int base = col * strideDw + ksDw + ((lane & 16) ? 4 : 0);
    v8i b;
    uint4 u0 = *(const uint4*)(lds + base);
    uint4 u1 = *(const uint4*)(lds + base + 8);
    b[0] = (int)u0.x; b[1] = (int)u0.y; b[2] = (int)u0.z; b[3] = (int)u0.w;
    b[4] = (int)u1.x; b[5] = (int)u1.y; b[6] = (int)u1.z; b[7] = (int)u1.w;
    return b;
}

#define LDS_STRIDE_DW 36

__global__ __launch_bounds__(256) __attribute__((amdgpu_waves_per_eu(2)))
void bitnet_gateup_i8_kernel(
    const signed char* __restrict__ x8, const signed char* __restrict__ wg8,
    const signed char* __restrict__ wu8, const float* __restrict__ scales,
    signed char* __restrict__ interq) {
    const int mBase = blockIdx.x * 128;
    const int nBase = blockIdx.y * 128;
    const int tid   = threadIdx.x;
    const int lane  = tid & 31;
    const int wid   = tid >> 5;
    const int mOff  = (wid & 3) * 32;
    const int nOff  = (wid >> 2) * 64;
    const int row   = tid >> 1;
    const int half  = tid & 1;

    const signed char* aRow = x8  + (size_t)(mBase + row) * HID + half * 64;
    const signed char* gRow = wg8 + (size_t)(nBase + row) * HID + half * 64;
    const signed char* uRow = wu8 + (size_t)(nBase + row) * HID + half * 64;

    v8i accG[2][4], accU[2][4];
#pragma unroll
    for (int i = 0; i < 2; ++i)
#pragma unroll
        for (int j = 0; j < 4; ++j) {
            accG[i][j] = (v8i){0, 0, 0, 0, 0, 0, 0, 0};
            accU[i][j] = (v8i){0, 0, 0, 0, 0, 0, 0, 0};
        }

    const int NK = HID / 128;

#if HAVE_ASYNC
    __shared__ uint As[2][128 * LDS_STRIDE_DW];
    __shared__ uint Bg[2][128 * LDS_STRIDE_DW];
    __shared__ uint Bu[2][128 * LDS_STRIDE_DW];
    const int ldsOff = row * LDS_STRIDE_DW + half * 16;

    auto issue = [&](int kb, int buf) {
        const char* ga = (const char*)aRow + kb * 128;
        const char* gg = (const char*)gRow + kb * 128;
        const char* gu = (const char*)uRow + kb * 128;
        char* la = (char*)&As[buf][ldsOff];
        char* lg = (char*)&Bg[buf][ldsOff];
        char* lu = (char*)&Bu[buf][ldsOff];
#pragma unroll
        for (int i = 0; i < 4; ++i) {
            async_cp_b128(ga + i * 16, la + i * 16);
            async_cp_b128(gg + i * 16, lg + i * 16);
            async_cp_b128(gu + i * 16, lu + i * 16);
        }
    };

    issue(0, 0);
    for (int kb = 0; kb < NK; ++kb) {
        if (kb + 1 < NK) { issue(kb + 1, (kb + 1) & 1); wait_async_12(); }
        else             { wait_async_0(); }
        __syncthreads();
        const uint* sA = As[kb & 1];
        const uint* sG = Bg[kb & 1];
        const uint* sU = Bu[kb & 1];
#else
    __shared__ uint AsS[128 * LDS_STRIDE_DW];
    __shared__ uint BgS[128 * LDS_STRIDE_DW];
    __shared__ uint BuS[128 * LDS_STRIDE_DW];
    uint* dA = &AsS[row * LDS_STRIDE_DW + half * 16];
    uint* dG = &BgS[row * LDS_STRIDE_DW + half * 16];
    uint* dU = &BuS[row * LDS_STRIDE_DW + half * 16];
    uint4 pa[4], pg[4], pu[4];
#pragma unroll
    for (int i = 0; i < 4; ++i) {
        pa[i] = ((const uint4*)aRow)[i];
        pg[i] = ((const uint4*)gRow)[i];
        pu[i] = ((const uint4*)uRow)[i];
    }
    for (int kb = 0; kb < NK; ++kb) {
#pragma unroll
        for (int i = 0; i < 4; ++i) {
            ((uint4*)dA)[i] = pa[i];
            ((uint4*)dG)[i] = pg[i];
            ((uint4*)dU)[i] = pu[i];
        }
        if (kb + 1 < NK) {
            const int k1 = (kb + 1) * 128;
#pragma unroll
            for (int i = 0; i < 4; ++i) {
                pa[i] = ((const uint4*)(aRow + k1))[i];
                pg[i] = ((const uint4*)(gRow + k1))[i];
                pu[i] = ((const uint4*)(uRow + k1))[i];
            }
        }
        __syncthreads();
        const uint* sA = AsS;
        const uint* sG = BgS;
        const uint* sU = BuS;
#endif
#pragma unroll
        for (int ks = 0; ks < 2; ++ks) {
            const int ksDw = ks * 16;
            v8i a0 = load_a_frag(sA, LDS_STRIDE_DW, mOff, ksDw, lane);
            v8i a1 = load_a_frag(sA, LDS_STRIDE_DW, mOff + 16, ksDw, lane);
#pragma unroll
            for (int fn = 0; fn < 4; ++fn) {
                v8i bg = load_b_frag(sG, LDS_STRIDE_DW, nOff + fn * 16, ksDw, lane);
                v8i bu = load_b_frag(sU, LDS_STRIDE_DW, nOff + fn * 16, ksDw, lane);
                accG[0][fn] = __builtin_amdgcn_wmma_i32_16x16x64_iu8(true, a0, true, bg, accG[0][fn], false, false);
                accG[1][fn] = __builtin_amdgcn_wmma_i32_16x16x64_iu8(true, a1, true, bg, accG[1][fn], false, false);
                accU[0][fn] = __builtin_amdgcn_wmma_i32_16x16x64_iu8(true, a0, true, bu, accU[0][fn], false, false);
                accU[1][fn] = __builtin_amdgcn_wmma_i32_16x16x64_iu8(true, a1, true, bu, accU[1][fn], false, false);
            }
        }
        __syncthreads();
    }

    const float sg128 = scales[0] * (1.0f / 128.0f);
    const float su128 = scales[1] * (1.0f / 128.0f);
#pragma unroll
    for (int fm = 0; fm < 2; ++fm) {
        const int m0 = mBase + mOff + fm * 16 + ((lane >> 4) << 3);
#pragma unroll
        for (int fn = 0; fn < 4; ++fn) {
            const int ncol = nBase + nOff + fn * 16 + (lane & 15);
#pragma unroll
            for (int r = 0; r < 8; ++r) {
                float g = (float)accG[fm][fn][r] * sg128;
                float u = (float)accU[fm][fn][r] * su128;
                float s = g / (1.0f + __expf(-g));
                float v = s * u * 128.0f;
                v = fminf(fmaxf(v, -128.0f), 127.0f);
                int q = (int)v;
                interq[(size_t)(m0 + r) * INTER + ncol] = (signed char)q;
            }
        }
    }
}

__global__ __launch_bounds__(256) __attribute__((amdgpu_waves_per_eu(2)))
void bitnet_down_i8_kernel(
    const signed char* __restrict__ interq, const signed char* __restrict__ wd8,
    const float* __restrict__ scales, float* __restrict__ out) {
    const int mBase = blockIdx.x * 128;
    const int nBase = blockIdx.y * 128;
    const int tid   = threadIdx.x;
    const int lane  = tid & 31;
    const int wid   = tid >> 5;
    const int mOff  = (wid & 3) * 32;
    const int nOff  = (wid >> 2) * 64;
    const int row   = tid >> 1;
    const int half  = tid & 1;

    const signed char* aRow = interq + (size_t)(mBase + row) * INTER + half * 64;
    const signed char* bRow = wd8    + (size_t)(nBase + row) * INTER + half * 64;

    v8i acc[2][4];
#pragma unroll
    for (int i = 0; i < 2; ++i)
#pragma unroll
        for (int j = 0; j < 4; ++j) acc[i][j] = (v8i){0, 0, 0, 0, 0, 0, 0, 0};

    const int NK = INTER / 128;

#if HAVE_ASYNC
    __shared__ uint As[2][128 * LDS_STRIDE_DW];
    __shared__ uint Bs[2][128 * LDS_STRIDE_DW];
    const int ldsOff = row * LDS_STRIDE_DW + half * 16;

    auto issue = [&](int kb, int buf) {
        const char* ga = (const char*)aRow + kb * 128;
        const char* gb = (const char*)bRow + kb * 128;
        char* la = (char*)&As[buf][ldsOff];
        char* lb = (char*)&Bs[buf][ldsOff];
#pragma unroll
        for (int i = 0; i < 4; ++i) {
            async_cp_b128(ga + i * 16, la + i * 16);
            async_cp_b128(gb + i * 16, lb + i * 16);
        }
    };

    issue(0, 0);
    for (int kb = 0; kb < NK; ++kb) {
        if (kb + 1 < NK) { issue(kb + 1, (kb + 1) & 1); wait_async_8(); }
        else             { wait_async_0(); }
        __syncthreads();
        const uint* sA = As[kb & 1];
        const uint* sB = Bs[kb & 1];
#else
    __shared__ uint AsS[128 * LDS_STRIDE_DW];
    __shared__ uint BsS[128 * LDS_STRIDE_DW];
    uint* dA = &AsS[row * LDS_STRIDE_DW + half * 16];
    uint* dB = &BsS[row * LDS_STRIDE_DW + half * 16];
    uint4 pa[4], pb[4];
#pragma unroll
    for (int i = 0; i < 4; ++i) {
        pa[i] = ((const uint4*)aRow)[i];
        pb[i] = ((const uint4*)bRow)[i];
    }
    for (int kb = 0; kb < NK; ++kb) {
#pragma unroll
        for (int i = 0; i < 4; ++i) {
            ((uint4*)dA)[i] = pa[i];
            ((uint4*)dB)[i] = pb[i];
        }
        if (kb + 1 < NK) {
            const int k1 = (kb + 1) * 128;
#pragma unroll
            for (int i = 0; i < 4; ++i) {
                pa[i] = ((const uint4*)(aRow + k1))[i];
                pb[i] = ((const uint4*)(bRow + k1))[i];
            }
        }
        __syncthreads();
        const uint* sA = AsS;
        const uint* sB = BsS;
#endif
#pragma unroll
        for (int ks = 0; ks < 2; ++ks) {
            const int ksDw = ks * 16;
            v8i a0 = load_a_frag(sA, LDS_STRIDE_DW, mOff, ksDw, lane);
            v8i a1 = load_a_frag(sA, LDS_STRIDE_DW, mOff + 16, ksDw, lane);
#pragma unroll
            for (int fn = 0; fn < 4; ++fn) {
                v8i b = load_b_frag(sB, LDS_STRIDE_DW, nOff + fn * 16, ksDw, lane);
                acc[0][fn] = __builtin_amdgcn_wmma_i32_16x16x64_iu8(true, a0, true, b, acc[0][fn], false, false);
                acc[1][fn] = __builtin_amdgcn_wmma_i32_16x16x64_iu8(true, a1, true, b, acc[1][fn], false, false);
            }
        }
        __syncthreads();
    }

    const float sd128 = scales[2] * 128.0f;
#pragma unroll
    for (int fm = 0; fm < 2; ++fm) {
        const int m0 = mBase + mOff + fm * 16 + ((lane >> 4) << 3);
#pragma unroll
        for (int fn = 0; fn < 4; ++fn) {
            const int ncol = nBase + nOff + fn * 16 + (lane & 15);
#pragma unroll
            for (int r = 0; r < 8; ++r) {
                float v = (float)acc[fm][fn][r] * sd128;
                v = fminf(fmaxf(v, -128.0f), 127.0f);
                int q = (int)v;
                out[(size_t)(m0 + r) * HID + ncol] = (float)q;
            }
        }
    }
}

extern "C" void kernel_launch(void* const* d_in, const int* in_sizes, int n_in,
                              void* d_out, int out_size, void* d_ws, size_t ws_size,
                              hipStream_t stream) {
    const float* x  = (const float*)d_in[0];
    const float* wg = (const float*)d_in[1];
    const float* wu = (const float*)d_in[2];
    const float* wd = (const float*)d_in[3];

    const size_t NX = (size_t)MTOT * HID;
    const size_t NW = (size_t)INTER * HID;

    char* ws = (char*)d_ws;
    float*       scales = (float*)ws;           ws += 256;
    signed char* x8     = (signed char*)ws;     ws += NX;
    signed char* wg8    = (signed char*)ws;     ws += NW;
    signed char* wu8    = (signed char*)ws;     ws += NW;
    signed char* wd8    = (signed char*)ws;     ws += NW;
    signed char* interq = (signed char*)ws;

    bitnet_maxabs_kernel<<<1, 256, 0, stream>>>(wg, scales + 0);
    bitnet_maxabs_kernel<<<1, 256, 0, stream>>>(wu, scales + 1);
    bitnet_maxabs_kernel<<<1, 256, 0, stream>>>(wd, scales + 2);

    bitnet_quant_x_kernel<<<(int)(NX / 4096), 256, 0, stream>>>(x, x8);
    bitnet_sign_kernel<<<(int)(NW / 4096), 256, 0, stream>>>(wg, wg8);
    bitnet_sign_kernel<<<(int)(NW / 4096), 256, 0, stream>>>(wu, wu8);
    bitnet_sign_kernel<<<(int)(NW / 4096), 256, 0, stream>>>(wd, wd8);

    dim3 gridGU(MTOT / 128, INTER / 128);
    bitnet_gateup_i8_kernel<<<gridGU, 256, 0, stream>>>(x8, wg8, wu8, scales, interq);

    dim3 gridD(MTOT / 128, HID / 128);
    bitnet_down_i8_kernel<<<gridD, 256, 0, stream>>>(interq, wd8, scales, (float*)d_out);
}
